// BottomGCN_79551384256723
// MI455X (gfx1250) — compile-verified, matrix operand order checked by reading, not run on hardware
//
#include <hip/hip_runtime.h>
#include <hip/hip_bf16.h>

#define N_NODES 200000
#define N_EDGES 600000
#define N_GRAPHS 4000
#define HIDDEN 128
#define DEPTH 4
#define IN_DIM 25
#define EDGE_DIM 11

typedef __attribute__((ext_vector_type(16))) _Float16 v16h;
typedef __attribute__((ext_vector_type(8)))  float    v8f;
typedef __attribute__((ext_vector_type(4)))  unsigned u32x4;

union FragCvt { v16h v; u32x4 q[2]; };

// CDNA5 16-bit A/B fragment loader. Per the ISA layout, lane L holds two
// contiguous 8-half runs at K = base..base+7 and base+16..base+23 where
// base = k0 + (L>=16 ? 8 : 0). Row (M for A, N for Bt) = provided index.
__device__ __forceinline__ v16h load_frag16(const _Float16* base_p, int row,
                                            int ldk, int k0, int lane) {
  const _Float16* p = base_p + row * ldk + k0 + ((lane & 16) ? 8 : 0);
  FragCvt u;
  u.q[0] = *(const u32x4*)(p);
  u.q[1] = *(const u32x4*)(p + 16);
  return u.v;
}

__device__ __forceinline__ float lrelu(float v) {
  return (v > 0.0f) ? v : 0.1f * v;
}

// ---------------- input projection: h = lrelu(x @ W_in + b_in) -> f16 -------
__global__ void __launch_bounds__(128) inproj_kernel(
    const float* __restrict__ x, const float* __restrict__ W,
    const float* __restrict__ b, _Float16* __restrict__ hH) {
  __shared__ float xs[IN_DIM];
  const int node = blockIdx.x;
  const int c = threadIdx.x;
  if (c < IN_DIM) xs[c] = x[(size_t)node * IN_DIM + c];
  __syncthreads();
  float acc = b[c];
#pragma unroll
  for (int k = 0; k < IN_DIM; ++k) acc += xs[k] * W[k * HIDDEN + c];
  acc = lrelu(acc);
  hH[(size_t)node * HIDDEN + c] = (_Float16)acc;
}

// ---- msg GEMM: lrelu(concat[h[src], edge_attr] @ Wmsg + b) scatter-add -----
// Block: 256 threads (8 waves), tile M=64 edges x N=128, K=139 padded to 160.
__global__ void __launch_bounds__(256) msg_kernel(
    const _Float16* __restrict__ hH, const float* __restrict__ edge_attr,
    const int* __restrict__ src, const int* __restrict__ dst,
    const float* __restrict__ W, const float* __restrict__ bias,
    float* __restrict__ aggr) {
  const int LDA = 168, LDB = 168;           // bank-conflict padded K strides
  __shared__ _Float16 sA[64 * 168];
  __shared__ _Float16 sB[128 * 168];        // B transposed: [N][K]
  __shared__ int sDst[64];

  const int tid = threadIdx.x;
  const int lane = tid & 31;
  const int wid = tid >> 5;
  const int wm = wid & 3, wn = wid >> 2;    // 4x2 wave grid of 16x64 subtiles
  const int m0 = wm * 16;

  // stage weights once per block (K padded 139 -> 160 with zeros)
  for (int p = tid; p < 128 * 160; p += 256) {
    const int k = p >> 7, n = p & 127;
    const float w = (k < 139) ? W[k * HIDDEN + n] : 0.0f;
    sB[n * LDB + k] = (_Float16)w;
  }

  const int nTiles = N_EDGES / 64;
  for (int tile = blockIdx.x; tile < nTiles; tile += gridDim.x) {
    const int e0 = tile * 64;
    {
      const int row = tid >> 2, q = tid & 3;
      const int e = e0 + row;
      const int s = src[e];
      const u32x4* hp = (const u32x4*)(hH + (size_t)s * HIDDEN) + q * 4;
      u32x4* dp = (u32x4*)(sA + row * LDA + q * 32);
      dp[0] = hp[0]; dp[1] = hp[1]; dp[2] = hp[2]; dp[3] = hp[3];
      if (q == 0) {
        sDst[row] = dst[e];
        const float* ea = edge_attr + (size_t)e * EDGE_DIM;
#pragma unroll
        for (int j = 0; j < EDGE_DIM; ++j)
          sA[row * LDA + HIDDEN + j] = (_Float16)ea[j];
#pragma unroll
        for (int j = HIDDEN + EDGE_DIM; j < 160; ++j)
          sA[row * LDA + j] = (_Float16)0.0f;
      }
    }
    __syncthreads();

    v8f acc[4];
#pragma unroll
    for (int j = 0; j < 4; ++j) { v8f z = {}; acc[j] = z; }

#pragma unroll
    for (int ks = 0; ks < 5; ++ks) {
      const int k0 = ks * 32;
      v16h a = load_frag16(sA, m0 + (lane & 15), LDA, k0, lane);
#pragma unroll
      for (int j = 0; j < 4; ++j) {
        v16h b = load_frag16(sB, wn * 64 + j * 16 + (lane & 15), LDB, k0, lane);
        acc[j] = __builtin_amdgcn_wmma_f32_16x16x32_f16(
            false, a, false, b, (short)0, acc[j], false, false);
      }
    }

    const int rbase = (lane & 16) ? 8 : 0;
#pragma unroll
    for (int j = 0; j < 4; ++j) {
      const int n = wn * 64 + j * 16 + (lane & 15);
      const float bv = bias[n];
#pragma unroll
      for (int r = 0; r < 8; ++r) {
        const float v = lrelu(acc[j][r] + bv);
        const int d = sDst[m0 + rbase + r];
        atomicAdd(&aggr[(size_t)d * HIDDEN + n], v);
      }
    }
    __syncthreads();
  }
}

// ---- update GEMM: relu(lrelu(concat[aggr, h] @ Wup + b)), + BN statistics --
// Tile M=64 nodes x N=128; K=256 done in two 128-wide phases to fit LDS.
__global__ void __launch_bounds__(256) update_kernel(
    const float* __restrict__ aggr, const _Float16* __restrict__ hH,
    const float* __restrict__ W, const float* __restrict__ bias,
    float* __restrict__ outBuf, float* __restrict__ gsum,
    float* __restrict__ gsumsq) {
  const int LDA = 136, LDB = 136;
  __shared__ _Float16 sA[64 * 136];
  __shared__ _Float16 sB[128 * 136];
  __shared__ float sSum[128], sSqr[128];

  const int tid = threadIdx.x;
  const int lane = tid & 31;
  const int wid = tid >> 5;
  const int wm = wid & 3, wn = wid >> 2;
  const int m0 = wm * 16;

  const int nTiles = N_NODES / 64;
  for (int tile = blockIdx.x; tile < nTiles; tile += gridDim.x) {
    if (tid < 128) { sSum[tid] = 0.0f; sSqr[tid] = 0.0f; }
    v8f acc[4];
#pragma unroll
    for (int j = 0; j < 4; ++j) { v8f z = {}; acc[j] = z; }

    const int node0 = tile * 64;
    for (int ph = 0; ph < 2; ++ph) {
      // stage weight K-slab (transposed)
      for (int p = tid; p < 128 * 128; p += 256) {
        const int k = p >> 7, n = p & 127;
        sB[n * LDB + k] = (_Float16)W[(size_t)(ph * 128 + k) * HIDDEN + n];
      }
      // stage A slab: phase0 = aggr (f32->f16), phase1 = h (already f16)
      const int row = tid >> 2, q = tid & 3;
      const int node = node0 + row;
      if (ph == 0) {
        const float* ap = aggr + (size_t)node * HIDDEN + q * 32;
        _Float16* dp = sA + row * LDA + q * 32;
#pragma unroll
        for (int i = 0; i < 32; ++i) dp[i] = (_Float16)ap[i];
      } else {
        const u32x4* hp = (const u32x4*)(hH + (size_t)node * HIDDEN) + q * 4;
        u32x4* dp = (u32x4*)(sA + row * LDA + q * 32);
        dp[0] = hp[0]; dp[1] = hp[1]; dp[2] = hp[2]; dp[3] = hp[3];
      }
      __syncthreads();

#pragma unroll
      for (int ks = 0; ks < 4; ++ks) {
        const int k0 = ks * 32;
        v16h a = load_frag16(sA, m0 + (lane & 15), LDA, k0, lane);
#pragma unroll
        for (int j = 0; j < 4; ++j) {
          v16h b =
              load_frag16(sB, wn * 64 + j * 16 + (lane & 15), LDB, k0, lane);
          acc[j] = __builtin_amdgcn_wmma_f32_16x16x32_f16(
              false, a, false, b, (short)0, acc[j], false, false);
        }
      }
      __syncthreads();
    }

    const int rbase = (lane & 16) ? 8 : 0;
#pragma unroll
    for (int j = 0; j < 4; ++j) {
      const int n = wn * 64 + j * 16 + (lane & 15);
      const float bv = bias[n];
      float ls = 0.0f, lq = 0.0f;
#pragma unroll
      for (int r = 0; r < 8; ++r) {
        float v = lrelu(acc[j][r] + bv);
        v = (v > 0.0f) ? v : 0.0f;  // relu
        outBuf[(size_t)(node0 + m0 + rbase + r) * HIDDEN + n] = v;
        ls += v;
        lq += v * v;
      }
      atomicAdd(&sSum[n], ls);
      atomicAdd(&sSqr[n], lq);
    }
    __syncthreads();
    if (tid < 128) {
      atomicAdd(&gsum[tid], sSum[tid]);
      atomicAdd(&gsumsq[tid], sSqr[tid]);
    }
    __syncthreads();
  }
}

// ------------- batch norm (in place) + refresh f16 mirror of h --------------
__global__ void __launch_bounds__(256) bn_kernel(
    const float* __restrict__ gsum, const float* __restrict__ gsumsq,
    const float* __restrict__ gam, const float* __restrict__ bet,
    float* __restrict__ nodeBuf, _Float16* __restrict__ hH) {
  const int idx = blockIdx.x * 256 + threadIdx.x;
  if (idx >= N_NODES * HIDDEN) return;
  const int c = idx & (HIDDEN - 1);
  const float inv_n = 1.0f / (float)N_NODES;
  const float mu = gsum[c] * inv_n;
  const float var = gsumsq[c] * inv_n - mu * mu;
  const float rstd = rsqrtf(var + 1e-5f);
  const float v = gam[c] * (nodeBuf[idx] - mu) * rstd + bet[c];
  nodeBuf[idx] = v;
  hH[idx] = (_Float16)v;
}

// ----------------------- global mean pool (scatter) -------------------------
__global__ void __launch_bounds__(256) pool_kernel(
    const float* __restrict__ nodeBuf, const int* __restrict__ batch,
    float* __restrict__ pool, float* __restrict__ cnt) {
  const int idx = blockIdx.x * 256 + threadIdx.x;
  if (idx >= N_NODES * HIDDEN) return;
  const int node = idx >> 7;
  const int c = idx & (HIDDEN - 1);
  const int g = batch[node];
  atomicAdd(&pool[(size_t)g * HIDDEN + c], nodeBuf[idx]);
  if (c == 0) atomicAdd(&cnt[g], 1.0f);
}

__global__ void __launch_bounds__(256) finalize_kernel(
    const float* __restrict__ pool, const float* __restrict__ cnt,
    float* __restrict__ out) {
  const int idx = blockIdx.x * 256 + threadIdx.x;
  if (idx >= N_GRAPHS * HIDDEN) return;
  const int g = idx >> 7;
  out[idx] = pool[idx] / fmaxf(cnt[g], 1.0f);
}

__global__ void __launch_bounds__(256) zero_kernel(float* __restrict__ p,
                                                   long long n) {
  long long i = (long long)blockIdx.x * blockDim.x + threadIdx.x;
  const long long stride = (long long)gridDim.x * blockDim.x;
  for (; i < n; i += stride) p[i] = 0.0f;
}

extern "C" void kernel_launch(void* const* d_in, const int* in_sizes, int n_in,
                              void* d_out, int out_size, void* d_ws,
                              size_t ws_size, hipStream_t stream) {
  (void)in_sizes; (void)n_in; (void)out_size; (void)ws_size;
  const float* x      = (const float*)d_in[0];
  const int*   ei     = (const int*)d_in[1];
  const float* eattr  = (const float*)d_in[2];
  const int*   batch  = (const int*)d_in[3];
  const float* W_in   = (const float*)d_in[4];
  const float* b_in   = (const float*)d_in[5];
  const float* W_msg  = (const float*)d_in[6];
  const float* b_msg  = (const float*)d_in[7];
  const float* W_up   = (const float*)d_in[8];
  const float* b_up   = (const float*)d_in[9];
  const float* gamma  = (const float*)d_in[10];
  const float* beta   = (const float*)d_in[11];

  const int* src = ei;
  const int* dst = ei + N_EDGES;

  // workspace carve (all 16B aligned). aggr+gsum+gsumsq and pool+cnt are
  // contiguous so each group clears with one zero_kernel call.
  char* ws = (char*)d_ws;
  float* nodeBuf = (float*)ws; ws += (size_t)N_NODES * HIDDEN * 4;
  float* aggr    = (float*)ws; ws += (size_t)N_NODES * HIDDEN * 4;
  float* gsum    = (float*)ws; ws += 128 * 4;
  float* gsumsq  = (float*)ws; ws += 128 * 4;
  float* pool    = (float*)ws; ws += (size_t)N_GRAPHS * HIDDEN * 4;
  float* cnt     = (float*)ws; ws += (size_t)N_GRAPHS * 4;
  _Float16* hH   = (_Float16*)ws; ws += (size_t)N_NODES * HIDDEN * 2;

  inproj_kernel<<<N_NODES, 128, 0, stream>>>(x, W_in, b_in, hH);

  for (int i = 0; i < DEPTH; ++i) {
    zero_kernel<<<2048, 256, 0, stream>>>(
        aggr, (long long)N_NODES * HIDDEN + 256);
    msg_kernel<<<1024, 256, 0, stream>>>(
        hH, eattr, src, dst, W_msg + (size_t)i * 139 * HIDDEN,
        b_msg + i * HIDDEN, aggr);
    update_kernel<<<1024, 256, 0, stream>>>(
        aggr, hH, W_up + (size_t)i * 256 * HIDDEN, b_up + i * HIDDEN, nodeBuf,
        gsum, gsumsq);
    bn_kernel<<<(N_NODES * HIDDEN) / 256, 256, 0, stream>>>(
        gsum, gsumsq, gamma + i * HIDDEN, beta + i * HIDDEN, nodeBuf, hH);
  }

  zero_kernel<<<512, 256, 0, stream>>>(
      pool, (long long)N_GRAPHS * HIDDEN + N_GRAPHS);
  pool_kernel<<<(N_NODES * HIDDEN) / 256, 256, 0, stream>>>(
      nodeBuf, batch, pool, cnt);
  finalize_kernel<<<(N_GRAPHS * HIDDEN) / 256, 256, 0, stream>>>(
      pool, cnt, (float*)d_out);
}
